// TimeVAEDecoder_16011638079794
// MI455X (gfx1250) — compile-verified, matrix operand order checked by reading, not run on hardware
//
#include <hip/hip_runtime.h>

// ---------------------------------------------------------------------------
// TimeVAE LSTM decoder for MI455X (gfx1250).
// B=1024, LATENT=128, H=256, F=64, SEQ=512, LAYERS=2.
// 64 persistent workgroups x 8 wave32; each WG owns a 16-row batch tile and
// runs the full 512-step recurrence with v_wmma_f32_16x16x32_bf16.
// Cell state stays in accumulator VGPRs; h round-trips through LDS as bf16.
// Weights are packed to bf16 (gate-permuted) once and re-streamed from L2
// every step. Pointers are laundered per-step (blocks LICM hoist/spill) and
// loads go through address_space(1) so they lower to global_load_b128.
// ---------------------------------------------------------------------------

typedef __bf16 bf16;
typedef __attribute__((ext_vector_type(16))) __bf16        v16bf;
typedef __attribute__((ext_vector_type(8)))  float         v8f;
typedef __attribute__((ext_vector_type(4)))  unsigned int  u32x4;   // builtin, AS-friendly
typedef __attribute__((address_space(1))) u32x4  gu4;
typedef __attribute__((address_space(1))) float  gfloat;

#define Bsz   1024
#define LAT   128
#define Hh    256
#define Ff    64
#define SEQn  512
#define G4H   1024   // 4*H

// packed gate row: p = wave*128 + gate*32 + j  <->  orig = gate*256 + wave*32 + j
__device__ __host__ inline int orig_row(int p) {
    int w = p >> 7, r = p & 127, g = r >> 5, j = r & 31;
    return g * 256 + w * 32 + j;
}

__device__ inline float sigmoidf_(float x) { return 1.0f / (1.0f + __expf(-x)); }

// opaque pointer launder: blocks LICM from hoisting loop-invariant weight
// loads out of the 512-step recurrence (which caused massive scratch spills)
__device__ inline unsigned long long launder_u64(const void* p) {
    unsigned long long u = (unsigned long long)p;
    asm volatile("" : "+v"(u));
    return u;
}

// ---- WMMA helpers ---------------------------------------------------------

__device__ inline v8f wmma_bf16(v16bf a, v16bf b, v8f c) {
    return __builtin_amdgcn_wmma_f32_16x16x32_bf16(
        /*neg_a=*/false, a, /*neg_b=*/false, b,
        /*c_mod=*/(short)0, c, /*reuse_a=*/false, /*reuse_b=*/false);
}

// 16x32 bf16 fragment: lane r=lane%16 holds row r; lanes<16 take K {0..7,16..23},
// lanes>=16 take K {8..15,24..31}  (row-major source, stride in elements).

// global (address_space(1)) fragment load -> global_load_b128 x2
__device__ inline v16bf load_tile_g(unsigned long long base_u, int stride,
                                    int row_base, int kk, unsigned lane) {
    int r  = row_base + (lane & 15);
    int ks = (lane & 16) ? 8 : 0;
    unsigned long long a = base_u + 2ull * (unsigned long long)(r * stride + kk + ks);
    union { u32x4 u[2]; v16bf f; } x;
    x.u[0] = *(const gu4*)a;
    x.u[1] = *(const gu4*)(a + 32);
    return x.f;
}

// LDS fragment load -> ds_load_b128 x2
__device__ inline v16bf load_tile_l(const bf16* base, int stride, int kk, unsigned lane) {
    int r  = lane & 15;
    int ks = (lane & 16) ? 8 : 0;
    const bf16* p = base + r * stride + kk + ks;
    union { u32x4 u[2]; v16bf f; } x;
    x.u[0] = *reinterpret_cast<const u32x4*>(p);
    x.u[1] = *reinterpret_cast<const u32x4*>(p + 16);
    return x.f;
}

// ---- precompute kernels ---------------------------------------------------

// initial h/c with the torch .view(num_layers,B,H) row-major reshape quirk
__global__ void k_init_state(const float* __restrict__ z,
                             const float* __restrict__ W_lh, const float* __restrict__ b_lh,
                             const float* __restrict__ W_lc, const float* __restrict__ b_lc,
                             float* __restrict__ h0i, float* __restrict__ c0i,
                             float* __restrict__ h1i, float* __restrict__ c1i) {
    int id = blockIdx.x * 256 + threadIdx.x;
    if (id >= 2 * Bsz * Hh) return;
    int l = id / (Bsz * Hh);
    int rem = id % (Bsz * Hh);
    int b = rem / Hh, h = rem % Hh;
    int zrow = l * (Bsz / 2) + (b >> 1);
    int col  = (b & 1) * Hh + h;
    const float* zr = z + zrow * LAT;
    const float* wh = W_lh + col * LAT;
    const float* wc = W_lc + col * LAT;
    float ah = b_lh[col], ac = b_lc[col];
    for (int k = 0; k < LAT; ++k) { float zv = zr[k]; ah += zv * wh[k]; ac += zv * wc[k]; }
    float* hd = l ? h1i : h0i;
    float* cd = l ? c1i : c0i;
    hd[b * Hh + h] = ah;
    cd[b * Hh + h] = ac;
}

__global__ void k_zproj(const float* __restrict__ z,
                        const float* __restrict__ W_lp, const float* __restrict__ b_lp,
                        float* __restrict__ zp) {
    int id = blockIdx.x * 256 + threadIdx.x;
    if (id >= Bsz * Hh) return;
    int b = id / Hh, h = id % Hh;
    const float* zr = z + b * LAT;
    const float* wr = W_lp + h * LAT;
    float acc = b_lp[h];
    for (int k = 0; k < LAT; ++k) acc += zr[k] * wr[k];
    zp[id] = acc;
}

// fused constant part of layer-0 gates: zproj@W_ih0[:,F:].T + b_ih0 + b_hh0,
// stored in gate-permuted column order [B][1024]
__global__ void k_bias0(const float* __restrict__ zp, const float* __restrict__ W_ih0,
                        const float* __restrict__ b_ih0, const float* __restrict__ b_hh0,
                        float* __restrict__ bias0) {
    int id = blockIdx.x * 256 + threadIdx.x;
    if (id >= Bsz * G4H) return;
    int b = id >> 10, p = id & 1023;
    int r = orig_row(p);
    const float* zr = zp + b * Hh;
    const float* wr = W_ih0 + r * (Ff + Hh) + Ff;
    float acc = b_ih0[r] + b_hh0[r];
    for (int k = 0; k < Hh; ++k) acc += zr[k] * wr[k];
    bias0[b * G4H + p] = acc;
}

__global__ void k_bias1(const float* __restrict__ b_ih1, const float* __restrict__ b_hh1,
                        float* __restrict__ bias1p) {
    int p = blockIdx.x * 256 + threadIdx.x;
    if (p >= G4H) return;
    int r = orig_row(p);
    bias1p[p] = b_ih1[r] + b_hh1[r];
}

// fp32 -> bf16 weight pack, optional gate-row permutation
__global__ void k_pack(const float* __restrict__ src, bf16* __restrict__ dst,
                       int rows, int cols, int sstride, int coff, int permute) {
    int id = blockIdx.x * 256 + threadIdx.x;
    if (id >= rows * cols) return;
    int r = id / cols, c = id % cols;
    int sr = permute ? orig_row(r) : r;
    dst[r * cols + c] = (bf16)src[sr * sstride + coff + c];
}

// ---- main persistent recurrence kernel ------------------------------------

__global__ __launch_bounds__(256)
void k_decoder(const float* __restrict__ h0i, const float* __restrict__ c0i,
               const float* __restrict__ h1i, const float* __restrict__ c1i,
               const float* __restrict__ bias0_,  // [B][1024] gate-permuted
               const float* __restrict__ bias1p,  // [1024]    gate-permuted
               const float* __restrict__ b_pre,   // [256]
               const float* __restrict__ b_out,   // [64]
               const bf16* __restrict__ Wp_in0_,  // [1024][64]
               const bf16* __restrict__ Wp_hh0_,  // [1024][256]
               const bf16* __restrict__ Wp_ih1_,  // [1024][256]
               const bf16* __restrict__ Wp_hh1_,  // [1024][256]
               const bf16* __restrict__ Wp_pre_,  // [256][256]
               const bf16* __restrict__ Wp_out_,  // [64][256]
               float* __restrict__ out)           // [B][SEQ][F]
{
    constexpr int XS = 72;    // sm_x row stride (16B-aligned rows)
    constexpr int HS = 272;   // h/s row stride
    __shared__ __align__(16) bf16 sm_x [16 * XS];
    __shared__ __align__(16) bf16 sm_h0[16 * HS];
    __shared__ __align__(16) bf16 sm_h1[16 * HS];
    __shared__ __align__(16) bf16 sm_s [16 * HS];

    const int wg      = blockIdx.x;      // 64 WGs, 16 batch rows each
    const int rowbase = wg * 16;
    const int tid     = threadIdx.x;
    const int w       = tid >> 5;        // wave 0..7: owns gate cols w*32..w*32+31
    const unsigned lane = tid & 31;
    const int N  = lane & 15;            // accumulator N within tile
    const int Mh = (lane >> 4) << 3;     // M half offset (0 or 8)

    // --- init LDS state ---
    for (int i = tid; i < 16 * Ff; i += 256)
        sm_x[(i >> 6) * XS + (i & 63)] = (bf16)0.0f;
    for (int i = tid; i < 16 * Hh; i += 256) {
        int r = i >> 8, c = i & 255;
        sm_h0[r * HS + c] = (bf16)h0i[(rowbase + r) * Hh + c];
        sm_h1[r * HS + c] = (bf16)h1i[(rowbase + r) * Hh + c];
        sm_s [r * HS + c] = (bf16)0.0f;
    }

    // cell state lives in registers for the whole rollout
    v8f c0a[2], c1a[2];
#pragma unroll
    for (int s = 0; s < 2; ++s)
#pragma unroll
        for (int v = 0; v < 8; ++v) {
            int M = v + Mh, col = w * 32 + s * 16 + N;
            c0a[s][v] = c0i[(rowbase + M) * Hh + col];
            c1a[s][v] = c1i[(rowbase + M) * Hh + col];
        }

    float b1r[8];
#pragma unroll
    for (int tt = 0; tt < 8; ++tt) b1r[tt] = bias1p[w * 128 + tt * 16 + N];
    float bpr[2];
#pragma unroll
    for (int s = 0; s < 2; ++s) bpr[s] = b_pre[w * 32 + s * 16 + N];
    float bor = (w < 4) ? b_out[w * 16 + N] : 0.0f;

    __syncthreads();

    for (int t = 0; t < SEQn; ++t) {
        // re-launder per step: weight loads must stay inside the loop and
        // stream from L2, not get hoisted + spilled to scratch
        unsigned long long bias0 = launder_u64(bias0_);
        unsigned long long Win0  = launder_u64(Wp_in0_);
        unsigned long long Whh0  = launder_u64(Wp_hh0_);
        unsigned long long Wih1  = launder_u64(Wp_ih1_);
        unsigned long long Whh1  = launder_u64(Wp_hh1_);
        unsigned long long Wpre  = launder_u64(Wp_pre_);
        unsigned long long Wout  = launder_u64(Wp_out_);

        // ---------------- layer 0 gates: [16 x 1024] ----------------
        v8f acc[8];
#pragma unroll
        for (int tt = 0; tt < 8; ++tt)
#pragma unroll
            for (int v = 0; v < 8; ++v)
                acc[tt][v] = ((const gfloat*)bias0)
                    [(rowbase + v + Mh) * G4H + w * 128 + tt * 16 + N];

        for (int kk = 0; kk < Ff; kk += 32) {               // dec_in part, K=64
            v16bf bfr[8];
#pragma unroll
            for (int tt = 0; tt < 8; ++tt)
                bfr[tt] = load_tile_g(Win0, Ff, w * 128 + tt * 16, kk, lane);
            v16bf a = load_tile_l(sm_x, XS, kk, lane);
#pragma unroll
            for (int tt = 0; tt < 8; ++tt) acc[tt] = wmma_bf16(a, bfr[tt], acc[tt]);
        }
        for (int kk = 0; kk < Hh; kk += 32) {               // h0 part, K=256
            v16bf bfr[8];
#pragma unroll
            for (int tt = 0; tt < 8; ++tt)
                bfr[tt] = load_tile_g(Whh0, Hh, w * 128 + tt * 16, kk, lane);
            v16bf a = load_tile_l(sm_h0, HS, kk, lane);
#pragma unroll
            for (int tt = 0; tt < 8; ++tt) acc[tt] = wmma_bf16(a, bfr[tt], acc[tt]);
        }
        __syncthreads();   // all reads of sm_x / sm_h0 done

        // cell update in registers; write new h0 to LDS as bf16
#pragma unroll
        for (int s = 0; s < 2; ++s)
#pragma unroll
            for (int v = 0; v < 8; ++v) {
                float ig = sigmoidf_(acc[0 + s][v]);
                float fg = sigmoidf_(acc[2 + s][v]);
                float gg = tanhf     (acc[4 + s][v]);
                float og = sigmoidf_(acc[6 + s][v]);
                float c  = fg * c0a[s][v] + ig * gg;
                c0a[s][v] = c;
                float h  = og * tanhf(c);
                sm_h0[(v + Mh) * HS + w * 32 + s * 16 + N] = (bf16)h;
            }
        __syncthreads();

        // ---------------- layer 1 gates: [16 x 1024] ----------------
#pragma unroll
        for (int tt = 0; tt < 8; ++tt)
#pragma unroll
            for (int v = 0; v < 8; ++v) acc[tt][v] = b1r[tt];

        for (int kk = 0; kk < Hh; kk += 32) {               // h0_new part
            v16bf bfr[8];
#pragma unroll
            for (int tt = 0; tt < 8; ++tt)
                bfr[tt] = load_tile_g(Wih1, Hh, w * 128 + tt * 16, kk, lane);
            v16bf a = load_tile_l(sm_h0, HS, kk, lane);
#pragma unroll
            for (int tt = 0; tt < 8; ++tt) acc[tt] = wmma_bf16(a, bfr[tt], acc[tt]);
        }
        for (int kk = 0; kk < Hh; kk += 32) {               // h1_old part
            v16bf bfr[8];
#pragma unroll
            for (int tt = 0; tt < 8; ++tt)
                bfr[tt] = load_tile_g(Whh1, Hh, w * 128 + tt * 16, kk, lane);
            v16bf a = load_tile_l(sm_h1, HS, kk, lane);
#pragma unroll
            for (int tt = 0; tt < 8; ++tt) acc[tt] = wmma_bf16(a, bfr[tt], acc[tt]);
        }
        __syncthreads();   // all reads of old sm_h1 done

        v8f h1r[2];
#pragma unroll
        for (int s = 0; s < 2; ++s)
#pragma unroll
            for (int v = 0; v < 8; ++v) {
                float ig = sigmoidf_(acc[0 + s][v]);
                float fg = sigmoidf_(acc[2 + s][v]);
                float gg = tanhf     (acc[4 + s][v]);
                float og = sigmoidf_(acc[6 + s][v]);
                float c  = fg * c1a[s][v] + ig * gg;
                c1a[s][v] = c;
                float h  = og * tanhf(c);
                h1r[s][v] = h;
                sm_h1[(v + Mh) * HS + w * 32 + s * 16 + N] = (bf16)h;
            }
        __syncthreads();

        // ---------------- pre: relu(h1@Wpre.T + b) + h1 -> sm_s ----------------
        v8f sacc[2];
#pragma unroll
        for (int s = 0; s < 2; ++s)
#pragma unroll
            for (int v = 0; v < 8; ++v) sacc[s][v] = bpr[s];

        for (int kk = 0; kk < Hh; kk += 32) {
            v16bf bfr[2];
#pragma unroll
            for (int s = 0; s < 2; ++s)
                bfr[s] = load_tile_g(Wpre, Hh, w * 32 + s * 16, kk, lane);
            v16bf a = load_tile_l(sm_h1, HS, kk, lane);
#pragma unroll
            for (int s = 0; s < 2; ++s) sacc[s] = wmma_bf16(a, bfr[s], sacc[s]);
        }
#pragma unroll
        for (int s = 0; s < 2; ++s)
#pragma unroll
            for (int v = 0; v < 8; ++v) {
                float sv = fmaxf(sacc[s][v], 0.0f) + h1r[s][v];
                sm_s[(v + Mh) * HS + w * 32 + s * 16 + N] = (bf16)sv;
            }
        __syncthreads();

        // ---------------- out: [16 x 64], waves 0..3 ----------------
        if (w < 4) {
            v8f oacc;
#pragma unroll
            for (int v = 0; v < 8; ++v) oacc[v] = bor;
            for (int kk = 0; kk < Hh; kk += 32) {
                v16bf b = load_tile_g(Wout, Hh, w * 16, kk, lane);
                v16bf a = load_tile_l(sm_s, HS, kk, lane);
                oacc = wmma_bf16(a, b, oacc);
            }
#pragma unroll
            for (int v = 0; v < 8; ++v) {
                float ov = oacc[v];
                // streamed once: bypass cache retention so weights stay hot in L2
                __builtin_nontemporal_store(
                    ov, &out[(size_t)(rowbase + v + Mh) * (SEQn * Ff) + t * Ff + w * 16 + N]);
                sm_x[(v + Mh) * XS + w * 16 + N] = (bf16)ov;   // next step's dec_in
            }
        }
        __syncthreads();
    }
}

// ---------------------------------------------------------------------------

extern "C" void kernel_launch(void* const* d_in, const int* in_sizes, int n_in,
                              void* d_out, int out_size, void* d_ws, size_t ws_size,
                              hipStream_t stream) {
    const float* z     = (const float*)d_in[0];
    const float* W_lh  = (const float*)d_in[1];
    const float* b_lh  = (const float*)d_in[2];
    const float* W_lc  = (const float*)d_in[3];
    const float* b_lc  = (const float*)d_in[4];
    const float* W_lp  = (const float*)d_in[5];
    const float* b_lp  = (const float*)d_in[6];
    const float* W_ih0 = (const float*)d_in[7];
    const float* b_ih0 = (const float*)d_in[8];
    const float* W_hh0 = (const float*)d_in[9];
    const float* b_hh0 = (const float*)d_in[10];
    const float* W_ih1 = (const float*)d_in[11];
    const float* b_ih1 = (const float*)d_in[12];
    const float* W_hh1 = (const float*)d_in[13];
    const float* b_hh1 = (const float*)d_in[14];
    const float* W_pre = (const float*)d_in[15];
    const float* b_pre = (const float*)d_in[16];
    const float* W_out = (const float*)d_in[17];
    const float* b_out = (const float*)d_in[18];
    float* out = (float*)d_out;

    char* ws = (char*)d_ws;
    size_t off = 0;
    auto take = [&](size_t bytes) -> char* {
        char* p = ws + off;
        off += (bytes + 255) & ~(size_t)255;
        return p;
    };
    const size_t SZf = (size_t)Bsz * Hh * sizeof(float);   // 1 MB
    float* h0i    = (float*)take(SZf);
    float* c0i    = (float*)take(SZf);
    float* h1i    = (float*)take(SZf);
    float* c1i    = (float*)take(SZf);
    float* zp     = (float*)take(SZf);
    float* bias0  = (float*)take((size_t)Bsz * G4H * sizeof(float));  // 4 MB
    float* bias1p = (float*)take(G4H * sizeof(float));
    bf16* Wp_in0  = (bf16*)take((size_t)G4H * Ff * sizeof(bf16));
    bf16* Wp_hh0  = (bf16*)take((size_t)G4H * Hh * sizeof(bf16));
    bf16* Wp_ih1  = (bf16*)take((size_t)G4H * Hh * sizeof(bf16));
    bf16* Wp_hh1  = (bf16*)take((size_t)G4H * Hh * sizeof(bf16));
    bf16* Wp_pre  = (bf16*)take((size_t)Hh * Hh * sizeof(bf16));
    bf16* Wp_out  = (bf16*)take((size_t)Ff * Hh * sizeof(bf16));

    // one-time (per launch) precompute + packing
    k_init_state<<<(2 * Bsz * Hh + 255) / 256, 256, 0, stream>>>(
        z, W_lh, b_lh, W_lc, b_lc, h0i, c0i, h1i, c1i);
    k_zproj<<<(Bsz * Hh + 255) / 256, 256, 0, stream>>>(z, W_lp, b_lp, zp);
    k_bias0<<<(Bsz * G4H + 255) / 256, 256, 0, stream>>>(zp, W_ih0, b_ih0, b_hh0, bias0);
    k_bias1<<<(G4H + 255) / 256, 256, 0, stream>>>(b_ih1, b_hh1, bias1p);

    k_pack<<<(G4H * Ff + 255) / 256, 256, 0, stream>>>(W_ih0, Wp_in0, G4H, Ff, Ff + Hh, 0, 1);
    k_pack<<<(G4H * Hh + 255) / 256, 256, 0, stream>>>(W_hh0, Wp_hh0, G4H, Hh, Hh, 0, 1);
    k_pack<<<(G4H * Hh + 255) / 256, 256, 0, stream>>>(W_ih1, Wp_ih1, G4H, Hh, Hh, 0, 1);
    k_pack<<<(G4H * Hh + 255) / 256, 256, 0, stream>>>(W_hh1, Wp_hh1, G4H, Hh, Hh, 0, 1);
    k_pack<<<(Hh * Hh + 255) / 256, 256, 0, stream>>>(W_pre, Wp_pre, Hh, Hh, Hh, 0, 0);
    k_pack<<<(Ff * Hh + 255) / 256, 256, 0, stream>>>(W_out, Wp_out, Ff, Hh, Hh, 0, 0);

    // persistent recurrence: 64 WGs x 8 waves, wave32
    k_decoder<<<Bsz / 16, 256, 0, stream>>>(
        h0i, c0i, h1i, c1i, bias0, bias1p, b_pre, b_out,
        Wp_in0, Wp_hh0, Wp_ih1, Wp_hh1, Wp_pre, Wp_out, out);
}
